// EMAU_45638322487448
// MI455X (gfx1250) — matrix operand order repaired and checked by reading, not compiled, not run on hardware
//
#include <hip/hip_runtime.h>
#include <hip/hip_bf16.h>

// ---------------------------------------------------------------------------
// EMAU (EM Attention Unit) for MI455X / gfx1250, wave32, WMMA bf16->f32,
// async global->LDS staging where layouts are copy-compatible.
// b=16, c=512, n=h*w=4096, k=64, 3 EM stages.
// ---------------------------------------------------------------------------

typedef __attribute__((ext_vector_type(16))) __bf16        v16bf;
typedef __attribute__((ext_vector_type(8)))  float         v8f;
typedef __attribute__((ext_vector_type(4)))  unsigned int  v4u;
typedef __attribute__((ext_vector_type(4)))  int           v4i;

#define BATCH 16
#define CDIM  512
#define NDIM  4096
#define KDIM  64
#define STAGES 3
#define LDB   40   // padded [N][K] B-tile stride (80B rows: 16B aligned, bank-spread)

#if __has_builtin(__builtin_amdgcn_global_load_async_to_lds_b128)
#define HAVE_ASYNC_LDS 1
#else
#define HAVE_ASYNC_LDS 0
#endif

__device__ __forceinline__ unsigned short f2bf(float f) {
    union { float f; unsigned u; } x; x.f = f;
    unsigned r = x.u + 0x7fffu + ((x.u >> 16) & 1u);   // round-to-nearest-even
    return (unsigned short)(r >> 16);
}

// 16B global -> LDS, async (ASYNCcnt) when available
__device__ __forceinline__ void g2l_b128(const unsigned short* g, unsigned short* l) {
#if HAVE_ASYNC_LDS
    __builtin_amdgcn_global_load_async_to_lds_b128(
        (__attribute__((address_space(1))) v4i*)(void*)const_cast<unsigned short*>(g),
        (__attribute__((address_space(3))) v4i*)(void*)l, 0, 0);
#else
    *(v4u*)l = *(const v4u*)g;
#endif
}
__device__ __forceinline__ void async_wait() {
#if HAVE_ASYNC_LDS
#if __has_builtin(__builtin_amdgcn_s_wait_asynccnt)
    __builtin_amdgcn_s_wait_asynccnt(0);
#else
    asm volatile("s_wait_asynccnt 0" ::: "memory");
#endif
#endif
}

// --- WMMA fragment builders (ISA 7.12.2, wave32, bf16 16x16x32) -------------
// A tile [M][K] in LDS: lane L -> row M=L%16; elements = two 16B runs.
__device__ __forceinline__ v16bf load_frag_a(const unsigned short* lds, int row, int ld) {
    int lane = threadIdx.x & 31;
    const unsigned short* p = lds + (row + (lane & 15)) * ld + (lane >> 4) * 8;
    union { v16bf v; v4u q[2]; } u;
    u.q[0] = *(const v4u*)(p);
    u.q[1] = *(const v4u*)(p + 16);
    return u.v;
}
// B tile stored transposed [N][K] in LDS: lane L -> col N=L%16; elements = two 16B runs.
__device__ __forceinline__ v16bf load_frag_bt(const unsigned short* lds, int col, int ld) {
    int lane = threadIdx.x & 31;
    const unsigned short* p = lds + (col + (lane & 15)) * ld + (lane >> 4) * 8;
    union { v16bf v; v4u q[2]; } u;
    u.q[0] = *(const v4u*)(p);
    u.q[1] = *(const v4u*)(p + 16);
    return u.v;
}

#define WMMA_BF16(a, b, c) \
    __builtin_amdgcn_wmma_f32_16x16x32_bf16(false, (a), false, (b), (short)0, (c), false, false)

// ---------------------------------------------------------------------------
// Elementwise converters
// ---------------------------------------------------------------------------
__global__ void k_cvt_bf16(const float* __restrict__ s, unsigned short* __restrict__ d, int n) {
    int i = blockIdx.x * blockDim.x + threadIdx.x;
    if (i < n) d[i] = f2bf(s[i]);
}
// broadcast mu[1,c,k] -> mu_b[b,c,k] and transposed mu_bT[b,k,c] (bf16)
__global__ void k_mu_init(const float* __restrict__ mu, unsigned short* __restrict__ d,
                          unsigned short* __restrict__ dt) {
    int i = blockIdx.x * blockDim.x + threadIdx.x;
    if (i < BATCH * CDIM * KDIM) {
        int b = i >> 15, rem = i & 32767, c = rem >> 6, k = rem & 63;
        unsigned short v = f2bf(mu[rem]);
        d[i] = v;
        dt[(b << 15) + (k << 9) + c] = v;
    }
}

// ---------------------------------------------------------------------------
// conv1: xf[b,o,n] = bf16( sum_c W1[o,c]*x[b,c,n] + bias[o] )
// grid (n/64, c/64, b), block 128 (4 waves), 64x64 tile, K=512 in 16 chunks
// ---------------------------------------------------------------------------
__global__ void k_conv1(const float* __restrict__ x, const unsigned short* __restrict__ w1,
                        const float* __restrict__ bias, unsigned short* __restrict__ xf) {
    __shared__ __align__(16) unsigned short ldsA[64 * 32];
    __shared__ __align__(16) unsigned short ldsBt[64 * LDB];
    const int tid = threadIdx.x, wave = tid >> 5;
    const int b = blockIdx.z, o0 = blockIdx.y * 64, n0 = blockIdx.x * 64;

    v8f acc[4] = {};
    for (int kc = 0; kc < 16; ++kc) {
        const int c0 = kc * 32;
#pragma unroll
        for (int it = 0; it < 2; ++it) {                     // A: W1 rows, async b128
            int idx = tid + it * 128, r = idx >> 2, q = idx & 3;
            g2l_b128(w1 + (o0 + r) * CDIM + c0 + q * 8, ldsA + r * 32 + q * 8);
        }
        for (int i = tid; i < 2048; i += 128) {              // Bt: x f32 -> bf16 [n][kk]
            int r = i >> 6, j = i & 63;
            ldsBt[j * LDB + r] = f2bf(x[(b * CDIM + c0 + r) * NDIM + n0 + j]);
        }
        async_wait();
        __syncthreads();
        v16bf a = load_frag_a(ldsA, 16 * wave, 32);
#pragma unroll
        for (int f = 0; f < 4; ++f) {
            v16bf bb = load_frag_bt(ldsBt, 16 * f, LDB);
            acc[f] = WMMA_BF16(a, bb, acc[f]);
        }
        __syncthreads();
    }
    const int lane = tid & 31, half = lane >> 4, cn = lane & 15;
#pragma unroll
    for (int f = 0; f < 4; ++f) {
        union { v8f v; float s[8]; } u; u.v = acc[f];
#pragma unroll
        for (int r = 0; r < 8; ++r) {
            int o = o0 + 16 * wave + r + 8 * half;
            int n = n0 + 16 * f + cn;
            xf[(b * CDIM + o) * NDIM + n] = f2bf(u.s[r] + bias[o]);
        }
    }
}

// ---------------------------------------------------------------------------
// z[b,n,k] = softmax_k( sum_c xf[b,c,n]*mu[b,c,k] ).  grid (n/64, b), block 128.
// A = xf^T (2-byte transpose scatter), B = muT rows (async b128).
// ---------------------------------------------------------------------------
__global__ void k_zstep(const unsigned short* __restrict__ xf,
                        const unsigned short* __restrict__ muT,
                        float* __restrict__ z) {
    __shared__ __align__(16) unsigned short ldsA[64 * 32];    // xf^T [n][kk]
    __shared__ __align__(16) unsigned short ldsBt[64 * LDB];  // muT  [k][kk]
    __shared__ float ldsC[64 * 64];
    __shared__ float rmax[64], rinv[64];
    const int tid = threadIdx.x, wave = tid >> 5;
    const int b = blockIdx.y, n0 = blockIdx.x * 64;
    const unsigned int* xfU = (const unsigned int*)xf;

    v8f acc[4] = {};
    for (int kc = 0; kc < 16; ++kc) {
        const int c0 = kc * 32;
#pragma unroll
        for (int it = 0; it < 2; ++it) {                     // Bt: muT rows, async b128
            int idx = tid + it * 128, col = idx >> 2, q = idx & 3;
            g2l_b128(muT + ((b * KDIM + col) << 9) + c0 + q * 8, ldsBt + col * LDB + q * 8);
        }
        for (int i = tid; i < 1024; i += 128) {              // A: transpose (2 n per uint)
            int kk = i >> 5, mp = i & 31;
            unsigned int v = xfU[(b * CDIM + c0 + kk) * (NDIM / 2) + (n0 >> 1) + mp];
            ldsA[(2 * mp + 0) * 32 + kk] = (unsigned short)(v & 0xffffu);
            ldsA[(2 * mp + 1) * 32 + kk] = (unsigned short)(v >> 16);
        }
        async_wait();
        __syncthreads();
        v16bf a = load_frag_a(ldsA, 16 * wave, 32);
#pragma unroll
        for (int f = 0; f < 4; ++f) {
            v16bf bb = load_frag_bt(ldsBt, 16 * f, LDB);
            acc[f] = WMMA_BF16(a, bb, acc[f]);
        }
        __syncthreads();
    }
    const int lane = tid & 31, half = lane >> 4, cn = lane & 15;
#pragma unroll
    for (int f = 0; f < 4; ++f) {
        union { v8f v; float s[8]; } u; u.v = acc[f];
#pragma unroll
        for (int r = 0; r < 8; ++r)
            ldsC[(16 * wave + r + 8 * half) * 64 + 16 * f + cn] = u.s[r];
    }
    __syncthreads();
    if (tid < 64) {                                // per-row softmax stats
        float m = -3.4e38f;
        for (int j = 0; j < 64; ++j) m = fmaxf(m, ldsC[tid * 64 + j]);
        float s = 0.f;
        for (int j = 0; j < 64; ++j) s += __expf(ldsC[tid * 64 + j] - m);
        rmax[tid] = m; rinv[tid] = 1.f / s;
    }
    __syncthreads();
    for (int i = tid; i < 4096; i += 128) {
        int row = i >> 6;
        z[((size_t)b * NDIM + n0 + row) * KDIM + (i & 63)] =
            __expf(ldsC[i] - rmax[row]) * rinv[row];
    }
}

// colsum[b,k] = sum_n z[b,n,k].  grid (b), block 256.
__global__ void k_colsum(const float* __restrict__ z, float* __restrict__ cs) {
    __shared__ float p[256];
    const int tid = threadIdx.x, b = blockIdx.x;
    const int k = tid & 63, s = tid >> 6;
    float a = 0.f;
    for (int n = s * 1024; n < s * 1024 + 1024; ++n)
        a += z[((size_t)b * NDIM + n) * KDIM + k];
    p[tid] = a;
    __syncthreads();
    if (tid < 64) cs[b * KDIM + tid] = p[tid] + p[64 + tid] + p[128 + tid] + p[192 + tid];
}

// mu_raw[b,c,k] = (sum_n xf[b,c,n]*z[b,n,k]) / (1e-6+colsum[b,k]).
// grid (c/64, b), block 128, K=n=4096 in 128 chunks of 32.
__global__ void k_mustep(const unsigned short* __restrict__ xf,
                         const float* __restrict__ z,
                         const float* __restrict__ cs,
                         float* __restrict__ mu_raw) {
    __shared__ __align__(16) unsigned short ldsA[64 * 32];    // xf [c][nn]
    __shared__ __align__(16) unsigned short ldsBt[64 * LDB];  // z  [k][nn]
    const int tid = threadIdx.x, wave = tid >> 5;
    const int b = blockIdx.y, c0 = blockIdx.x * 64;

    v8f acc[4] = {};
    for (int kc = 0; kc < 128; ++kc) {
        const int nk = kc * 32;
#pragma unroll
        for (int it = 0; it < 2; ++it) {                     // A: xf rows, async b128
            int idx = tid + it * 128, r = idx >> 2, q = idx & 3;
            g2l_b128(xf + (b * CDIM + c0 + r) * NDIM + nk + q * 8, ldsA + r * 32 + q * 8);
        }
        for (int i = tid; i < 2048; i += 128) {              // Bt: z f32 -> bf16 [k][nn]
            int kk = i >> 6, col = i & 63;
            ldsBt[col * LDB + kk] = f2bf(z[((size_t)b * NDIM + nk + kk) * KDIM + col]);
        }
        async_wait();
        __syncthreads();
        v16bf a = load_frag_a(ldsA, 16 * wave, 32);
#pragma unroll
        for (int f = 0; f < 4; ++f) {
            v16bf bb = load_frag_bt(ldsBt, 16 * f, LDB);
            acc[f] = WMMA_BF16(a, bb, acc[f]);
        }
        __syncthreads();
    }
    const int lane = tid & 31, half = lane >> 4, cn = lane & 15;
#pragma unroll
    for (int f = 0; f < 4; ++f) {
        int colk = 16 * f + cn;
        float sc = 1.f / (1e-6f + cs[b * KDIM + colk]);
        union { v8f v; float s[8]; } u; u.v = acc[f];
#pragma unroll
        for (int r = 0; r < 8; ++r) {
            int c = c0 + 16 * wave + r + 8 * half;
            mu_raw[(b * CDIM + c) * KDIM + colk] = u.s[r] * sc;
        }
    }
}

// mu/muT (bf16) = mu_raw / (1e-6 + ||mu_raw||_2 over c).  grid (b), block 256.
__global__ void k_munorm(const float* __restrict__ mu_raw, unsigned short* __restrict__ mu,
                         unsigned short* __restrict__ muT) {
    __shared__ float p[256];
    __shared__ float scl[64];
    const int tid = threadIdx.x, b = blockIdx.x;
    const int k = tid & 63, s = tid >> 6;
    float a = 0.f;
    for (int c = s * 128; c < s * 128 + 128; ++c) {
        float v = mu_raw[(b * CDIM + c) * KDIM + k];
        a += v * v;
    }
    p[tid] = a;
    __syncthreads();
    if (tid < 64) {
        float ss = p[tid] + p[64 + tid] + p[128 + tid] + p[192 + tid];
        scl[tid] = 1.f / (1e-6f + sqrtf(ss));
    }
    __syncthreads();
    for (int i = tid; i < CDIM * KDIM; i += 256) {
        int c = i >> 6, kk = i & 63;
        unsigned short v = f2bf(mu_raw[(b << 15) + i] * scl[kk]);
        mu[(b << 15) + i] = v;
        muT[(b << 15) + (kk << 9) + c] = v;
    }
}

// xr[b,c,n] = bf16( relu( sum_k mu[b,c,k]*z[b,n,k] ) ).
// grid (n/64, c/64, b), block 128, K=64 in 2 chunks of 32.
__global__ void k_xr(const unsigned short* __restrict__ mu, const float* __restrict__ z,
                     unsigned short* __restrict__ xr) {
    __shared__ __align__(16) unsigned short ldsM[64 * 64];   // mu tile [c][k], async
    __shared__ __align__(16) unsigned short ldsZ[64 * 72];   // z tile  [n][k]
    const int tid = threadIdx.x, wave = tid >> 5;
    const int b = blockIdx.z, c0 = blockIdx.y * 64, n0 = blockIdx.x * 64;

#pragma unroll
    for (int it = 0; it < 4; ++it) {                         // 8KB contiguous, async b128
        int idx = tid + it * 128;
        g2l_b128(mu + ((b * CDIM + c0) << 6) + idx * 8, ldsM + idx * 8);
    }
    for (int i = tid; i < 4096; i += 128) {
        int j = i >> 6, k = i & 63;
        ldsZ[j * 72 + k] = f2bf(z[((size_t)b * NDIM + n0 + j) * KDIM + k]);
    }
    async_wait();
    __syncthreads();

    v8f acc[4] = {};
#pragma unroll
    for (int kc = 0; kc < 2; ++kc) {
        const int kk0 = kc * 32;
        v16bf a = load_frag_a(ldsM + kk0, 16 * wave, 64);
#pragma unroll
        for (int f = 0; f < 4; ++f) {
            v16bf bb = load_frag_bt(ldsZ + kk0, 16 * f, 72);
            acc[f] = WMMA_BF16(a, bb, acc[f]);
        }
    }
    const int lane = tid & 31, half = lane >> 4, cn = lane & 15;
#pragma unroll
    for (int f = 0; f < 4; ++f) {
        union { v8f v; float s[8]; } u; u.v = acc[f];
#pragma unroll
        for (int r = 0; r < 8; ++r) {
            int c = c0 + 16 * wave + r + 8 * half;
            int n = n0 + 16 * f + cn;
            xr[(b * CDIM + c) * NDIM + n] = f2bf(fmaxf(u.s[r], 0.f));
        }
    }
}

// out[b,o,n] = relu( (sum_c W2[o,c]*xr[b,c,n]) * inv[o] + (beta[o]-mean[o]*inv[o]) + x[b,o,n] )
// grid (n/64, o/64, b), block 128.
__global__ void k_conv2(const unsigned short* __restrict__ xr, const unsigned short* __restrict__ w2,
                        const float* __restrict__ gamma, const float* __restrict__ beta,
                        const float* __restrict__ mean, const float* __restrict__ var,
                        const float* __restrict__ x, float* __restrict__ out) {
    __shared__ __align__(16) unsigned short ldsA[64 * 32];
    __shared__ __align__(16) unsigned short ldsBt[64 * LDB];
    const int tid = threadIdx.x, wave = tid >> 5;
    const int b = blockIdx.z, o0 = blockIdx.y * 64, n0 = blockIdx.x * 64;
    const unsigned int* xrU = (const unsigned int*)xr;

    v8f acc[4] = {};
    for (int kc = 0; kc < 16; ++kc) {
        const int c0 = kc * 32;
#pragma unroll
        for (int it = 0; it < 2; ++it) {                     // A: W2 rows, async b128
            int idx = tid + it * 128, r = idx >> 2, q = idx & 3;
            g2l_b128(w2 + (o0 + r) * CDIM + c0 + q * 8, ldsA + r * 32 + q * 8);
        }
        for (int i = tid; i < 1024; i += 128) {              // Bt: xr transpose (2 n/uint)
            int r = i >> 5, jp = i & 31;
            unsigned int v = xrU[(b * CDIM + c0 + r) * (NDIM / 2) + (n0 >> 1) + jp];
            ldsBt[(2 * jp + 0) * LDB + r] = (unsigned short)(v & 0xffffu);
            ldsBt[(2 * jp + 1) * LDB + r] = (unsigned short)(v >> 16);
        }
        async_wait();
        __syncthreads();
        v16bf a = load_frag_a(ldsA, 16 * wave, 32);
#pragma unroll
        for (int f = 0; f < 4; ++f) {
            v16bf bb = load_frag_bt(ldsBt, 16 * f, LDB);
            acc[f] = WMMA_BF16(a, bb, acc[f]);
        }
        __syncthreads();
    }
    const int lane = tid & 31, half = lane >> 4, cn = lane & 15;
#pragma unroll
    for (int f = 0; f < 4; ++f) {
        union { v8f v; float s[8]; } u; u.v = acc[f];
#pragma unroll
        for (int r = 0; r < 8; ++r) {
            int o = o0 + 16 * wave + r + 8 * half;
            int n = n0 + 16 * f + cn;
            float inv = gamma[o] * rsqrtf(var[o] + 1e-5f);
            float v = u.s[r] * inv + (beta[o] - mean[o] * inv) + x[(b * CDIM + o) * NDIM + n];
            out[(b * CDIM + o) * NDIM + n] = fmaxf(v, 0.f);
        }
    }
}

// ---------------------------------------------------------------------------
extern "C" void kernel_launch(void* const* d_in, const int* in_sizes, int n_in,
                              void* d_out, int out_size, void* d_ws, size_t ws_size,
                              hipStream_t stream) {
    const float* x      = (const float*)d_in[0];
    const float* w1     = (const float*)d_in[1];
    const float* bias   = (const float*)d_in[2];
    const float* mu0    = (const float*)d_in[3];
    const float* w2     = (const float*)d_in[4];
    const float* gamma  = (const float*)d_in[5];
    const float* beta   = (const float*)d_in[6];
    const float* mean   = (const float*)d_in[7];
    const float* var    = (const float*)d_in[8];
    float* out = (float*)d_out;

    char* ws = (char*)d_ws;
    size_t off = 0;
    auto alloc = [&](size_t bytes) { void* p = ws + off; off = (off + bytes + 255) & ~(size_t)255; return p; };
    unsigned short* w1b  = (unsigned short*)alloc((size_t)CDIM * CDIM * 2);          // 0.5 MB
    unsigned short* w2b  = (unsigned short*)alloc((size_t)CDIM * CDIM * 2);          // 0.5 MB
    unsigned short* mub  = (unsigned short*)alloc((size_t)BATCH * CDIM * KDIM * 2);  // 1 MB
    unsigned short* mubT = (unsigned short*)alloc((size_t)BATCH * CDIM * KDIM * 2);  // 1 MB
    unsigned short* xfb  = (unsigned short*)alloc((size_t)BATCH * CDIM * NDIM * 2);  // 64 MB
    float*          zbuf = (float*)alloc((size_t)BATCH * NDIM * KDIM * 4);           // 16 MB
    float*          csum = (float*)alloc((size_t)BATCH * KDIM * 4);
    float*          mur  = (float*)alloc((size_t)BATCH * CDIM * KDIM * 4);           // 2 MB
    unsigned short* xrb  = (unsigned short*)alloc((size_t)BATCH * CDIM * NDIM * 2);  // 64 MB
    (void)ws_size; (void)n_in; (void)in_sizes; (void)out_size;

    // weight / mu conversion
    k_cvt_bf16<<<(CDIM * CDIM + 255) / 256, 256, 0, stream>>>(w1, w1b, CDIM * CDIM);
    k_cvt_bf16<<<(CDIM * CDIM + 255) / 256, 256, 0, stream>>>(w2, w2b, CDIM * CDIM);
    k_mu_init<<<(BATCH * CDIM * KDIM + 255) / 256, 256, 0, stream>>>(mu0, mub, mubT);

    // conv1
    k_conv1<<<dim3(NDIM / 64, CDIM / 64, BATCH), 128, 0, stream>>>(x, w1b, bias, xfb);

    // 3 EM stages
    for (int s = 0; s < STAGES; ++s) {
        k_zstep<<<dim3(NDIM / 64, BATCH), 128, 0, stream>>>(xfb, mubT, zbuf);
        k_colsum<<<BATCH, 256, 0, stream>>>(zbuf, csum);
        k_mustep<<<dim3(CDIM / 64, BATCH), 128, 0, stream>>>(xfb, zbuf, csum, mur);
        k_munorm<<<BATCH, 256, 0, stream>>>(mur, mub, mubT);
    }

    // reconstruction + conv2/BN/residual
    k_xr<<<dim3(NDIM / 64, CDIM / 64, BATCH), 128, 0, stream>>>(mub, zbuf, xrb);
    k_conv2<<<dim3(NDIM / 64, CDIM / 64, BATCH), 128, 0, stream>>>(xrb, w2b, gamma, beta, mean, var, x, out);
}
